// Generator_67388036874295
// MI455X (gfx1250) — matrix operand order repaired and checked by reading, not compiled, not run on hardware
//
#include <hip/hip_runtime.h>
#include <hip/hip_bf16.h>
#include <math.h>

// Problem constants (from reference)
#define T_STEPS 2048
#define NB      32
#define LAT     128
#define EMB_D   128
#define KDIM    256      // EMB+LAT == H == 256 (GEMM K)
#define HID     256
#define V_OUT   1000

// Kernel organization
#define ROWS    16       // batch rows per workgroup (2 workgroups total)
#define BLOCK   512      // 16 waves of 32
#define NWAVES  16

#define GATE_N  768      // packed i|g|o gate columns (f gate is dead: c_prev==0)
#define FC_N    1008     // V padded to 16
#define AP      264      // abuf row stride in bf16 (256 + 8 pad -> bank spread)
#define GP      776      // gbuf row stride in f32
#define LP      1016     // lbuf row stride in f32

typedef __attribute__((ext_vector_type(16))) __bf16 v16bf;
typedef __attribute__((ext_vector_type(8)))  float  v8f;
typedef __attribute__((ext_vector_type(4)))  unsigned int u32x4;

union Frag16 { v16bf v; u32x4 q[2]; };

__device__ __forceinline__ unsigned short f2bf(float f) {
  union { float f; unsigned u; } c; c.f = f;
  unsigned u = c.u;
  u += 0x7FFFu + ((u >> 16) & 1u);   // round-to-nearest-even
  return (unsigned short)(u >> 16);
}

__device__ __forceinline__ float sigm(float x) { return 1.0f / (1.0f + __expf(-x)); }

__device__ __forceinline__ float hashu01(unsigned a, unsigned b, unsigned c) {
  unsigned h = a * 0x9E3779B1u ^ b * 0x85EBCA77u ^ c * 0xC2B2AE3Du;
  h ^= h >> 16; h *= 0x7FEB352Du; h ^= h >> 15; h *= 0x846CA68Bu; h ^= h >> 16;
  return (float)(h >> 8) * (1.0f / 16777216.0f);
}

// ---------------------------------------------------------------------------
// Prep: pack weights to bf16, drop dead W_hh and dead f-gate, fuse biases.
// Packed gate rows p: [0,256)=i -> src p; [256,512)=g -> src p+256 (rows 512..768);
// [512,768)=o -> src p+256 (rows 768..1024).  i.e. src = p<256 ? p : p+256.
// ---------------------------------------------------------------------------
__global__ void prep_pack(const float* __restrict__ W0, const float* __restrict__ W1,
                          const float* __restrict__ W2,
                          const float* __restrict__ bi0, const float* __restrict__ bh0,
                          const float* __restrict__ bi1, const float* __restrict__ bh1,
                          const float* __restrict__ bi2, const float* __restrict__ bh2,
                          const float* __restrict__ Wfc, const float* __restrict__ bfc,
                          unsigned short* __restrict__ wpk, unsigned short* __restrict__ wfcp,
                          float* __restrict__ bsum, float* __restrict__ bfcp) {
  const int GATES = 3 * GATE_N * KDIM;
  const int FCE   = FC_N * KDIM;
  int i = blockIdx.x * blockDim.x + threadIdx.x;
  int total = GATES + FCE + 3 * GATE_N + FC_N;
  if (i >= total) return;
  if (i < GATES) {
    int L = i / (GATE_N * KDIM);
    int r = i - L * GATE_N * KDIM;
    int p = r / KDIM, k = r - p * KDIM;
    int src = (p < 256) ? p : p + 256;
    const float* W = (L == 0) ? W0 : (L == 1) ? W1 : W2;
    wpk[i] = f2bf(W[src * KDIM + k]);
  } else if (i < GATES + FCE) {
    int r = i - GATES;
    int p = r / KDIM;
    wfcp[r] = (p < V_OUT) ? f2bf(Wfc[r]) : (unsigned short)0;
  } else if (i < GATES + FCE + 3 * GATE_N) {
    int r = i - GATES - FCE;
    int L = r / GATE_N;
    int p = r - L * GATE_N;
    int src = (p < 256) ? p : p + 256;
    const float* bi = (L == 0) ? bi0 : (L == 1) ? bi1 : bi2;
    const float* bh = (L == 0) ? bh0 : (L == 1) ? bh1 : bh2;
    bsum[r] = bi[src] + bh[src];
  } else {
    int p = i - GATES - FCE - 3 * GATE_N;
    bfcp[p] = (p < V_OUT) ? bfc[p] : -1e30f;
  }
}

// ---------------------------------------------------------------------------
// Persistent generator: one workgroup owns 16 batch rows for all 2048 steps.
// ---------------------------------------------------------------------------
__global__ void __launch_bounds__(BLOCK, 1)
lstm_gen(const float* __restrict__ x, const float* __restrict__ emb,
         const unsigned short* __restrict__ wpk, const unsigned short* __restrict__ wfcp,
         const float* __restrict__ bsum, const float* __restrict__ bfcp,
         float* __restrict__ out) {
  extern __shared__ char smem[];
  unsigned short* abuf = (unsigned short*)smem;                           // ROWS x AP bf16
  float* gbuf = (float*)(smem + ROWS * AP * 2);                           // ROWS x GP f32
  float* lbuf = (float*)(smem + ROWS * AP * 2 + ROWS * GP * 4);           // ROWS x LP f32
  int* tokbuf = (int*)(smem + ROWS * AP * 2 + (ROWS * GP + ROWS * LP) * 4);

  const int tid  = threadIdx.x;
  const int wave = tid >> 5;
  const int lane = tid & 31;
  const int half = lane >> 4;     // selects K-sub-range per WMMA 16-bit layout
  const int laneN = lane & 15;    // N (B/C/D) column, and A row (M)
  const int row0 = blockIdx.x * ROWS;

  // inp(0) = concat(x[:,0,:], zeros)
  for (int i = tid; i < ROWS * KDIM; i += BLOCK) {
    int r = i >> 8, j = i & 255;
    float v = (j < LAT) ? x[((size_t)(row0 + r) * T_STEPS) * LAT + j] : 0.0f;
    abuf[r * AP + j] = f2bf(v);
  }

  #pragma unroll 1
  for (int t = 0; t < T_STEPS; ++t) {
    // ----- three zero-state LSTM cells -----
    #pragma unroll 1
    for (int L = 0; L < 3; ++L) {
      __syncthreads();   // abuf (A matrix) ready
      const unsigned short* Wl = wpk + (size_t)L * GATE_N * KDIM;
      const float* bl = bsum + L * GATE_N;
      const unsigned short* arow = abuf + laneN * AP;
      for (int tn = wave; tn < GATE_N / 16; tn += NWAVES) {
        v8f acc = {};
        const unsigned short* wrow = Wl + (size_t)(tn * 16 + laneN) * KDIM;
        #pragma unroll
        for (int ks = 0; ks < 8; ++ks) {
          Frag16 a, b;
          // A 16x32 bf16: elem j -> K = 32ks + 16*half + j
          a.q[0] = *(const u32x4*)(arow + ks * 32 + half * 16);
          a.q[1] = *(const u32x4*)(arow + ks * 32 + half * 16 + 8);
          // B 32x16 bf16: elem j -> K = 32ks + 16*half + j, N = laneN (row-major W)
          const u32x4* bp = (const u32x4*)(wrow + ks * 32 + half * 16);
          b.q[0] = bp[0]; b.q[1] = bp[1];
          acc = __builtin_amdgcn_wmma_f32_16x16x32_bf16(
              false, a.v, false, b.v, (short)0, acc, false, false);
        }
        float bias = bl[tn * 16 + laneN];
        #pragma unroll
        for (int j = 0; j < 8; ++j)   // C elem j: M = j + 8*half, N = laneN
          gbuf[(j + half * 8) * GP + tn * 16 + laneN] = acc[j] + bias;
      }
      __syncthreads();
      // h = sigmoid(o) * tanh(sigmoid(i) * tanh(g));   f*c_prev == 0
      for (int i = tid; i < ROWS * HID; i += BLOCK) {
        int r = i >> 8, j = i & 255;
        float gi = gbuf[r * GP + j];
        float gg = gbuf[r * GP + 256 + j];
        float go = gbuf[r * GP + 512 + j];
        float cc = sigm(gi) * tanhf(gg);
        abuf[r * AP + j] = f2bf(sigm(go) * tanhf(cc));
      }
    }
    __syncthreads();
    // ----- FC: logits = h2 @ W_fc^T + b_fc -----
    {
      const unsigned short* arow = abuf + laneN * AP;
      for (int tn = wave; tn < FC_N / 16; tn += NWAVES) {
        v8f acc = {};
        const unsigned short* wrow = wfcp + (size_t)(tn * 16 + laneN) * KDIM;
        #pragma unroll
        for (int ks = 0; ks < 8; ++ks) {
          Frag16 a, b;
          a.q[0] = *(const u32x4*)(arow + ks * 32 + half * 16);
          a.q[1] = *(const u32x4*)(arow + ks * 32 + half * 16 + 8);
          const u32x4* bp = (const u32x4*)(wrow + ks * 32 + half * 16);
          b.q[0] = bp[0]; b.q[1] = bp[1];
          acc = __builtin_amdgcn_wmma_f32_16x16x32_bf16(
              false, a.v, false, b.v, (short)0, acc, false, false);
        }
        float bias = bfcp[tn * 16 + laneN];
        #pragma unroll
        for (int j = 0; j < 8; ++j)
          lbuf[(j + half * 8) * LP + tn * 16 + laneN] = acc[j] + bias;
      }
    }
    __syncthreads();
    // ----- softmax -> out, Gumbel-max sample -> tokbuf  (wave == row) -----
    {
      const int r = wave;
      const int bg = row0 + r;
      const float* lr = lbuf + r * LP;
      float mx = -3.4e38f;
      for (int v = lane; v < V_OUT; v += 32) mx = fmaxf(mx, lr[v]);
      for (int o = 16; o; o >>= 1) mx = fmaxf(mx, __shfl_xor(mx, o, 32));
      float s = 0.0f, gbest = -3.4e38f; int ibest = 0;
      for (int v = lane; v < V_OUT; v += 32) {
        float l = lr[v];
        s += __expf(l - mx);
        float u = hashu01((unsigned)t, (unsigned)bg, (unsigned)v);
        float z = l - __logf(-__logf(u * 0.9999999f + 1e-9f));   // Gumbel-max
        if (z > gbest) { gbest = z; ibest = v; }
      }
      for (int o = 16; o; o >>= 1) s += __shfl_xor(s, o, 32);
      for (int o = 16; o; o >>= 1) {
        float ov = __shfl_xor(gbest, o, 32);
        int   oi = __shfl_xor(ibest, o, 32);
        if (ov > gbest || (ov == gbest && oi < ibest)) { gbest = ov; ibest = oi; }
      }
      float inv = 1.0f / s;
      float* orow = out + ((size_t)bg * T_STEPS + t) * V_OUT;
      for (int v = lane; v < V_OUT; v += 32) orow[v] = __expf(lr[v] - mx) * inv;
      if (lane == 0) tokbuf[r] = ibest;
    }
    __syncthreads();
    // ----- next input: concat(x[:,t+1,:], emb[tok]) -----
    if (t + 1 < T_STEPS) {
      for (int i = tid; i < ROWS * KDIM; i += BLOCK) {
        int r = i >> 8, j = i & 255;
        float v = (j < LAT)
            ? x[((size_t)(row0 + r) * T_STEPS + (t + 1)) * LAT + j]
            : emb[tokbuf[r] * EMB_D + (j - LAT)];
        abuf[r * AP + j] = f2bf(v);
      }
    }
  }
}

extern "C" void kernel_launch(void* const* d_in, const int* in_sizes, int n_in,
                              void* d_out, int out_size, void* d_ws, size_t ws_size,
                              hipStream_t stream) {
  (void)in_sizes; (void)n_in; (void)out_size; (void)ws_size;
  const float* x   = (const float*)d_in[0];
  const float* emb = (const float*)d_in[1];
  const float* W0  = (const float*)d_in[2];
  const float* bi0 = (const float*)d_in[4];
  const float* bh0 = (const float*)d_in[5];
  const float* W1  = (const float*)d_in[6];
  const float* bi1 = (const float*)d_in[8];
  const float* bh1 = (const float*)d_in[9];
  const float* W2  = (const float*)d_in[10];
  const float* bi2 = (const float*)d_in[12];
  const float* bh2 = (const float*)d_in[13];
  const float* Wfc = (const float*)d_in[14];
  const float* bfc = (const float*)d_in[15];
  float* out = (float*)d_out;

  char* ws = (char*)d_ws;
  unsigned short* wpk  = (unsigned short*)ws;                               // 3*768*256 bf16
  unsigned short* wfcp = (unsigned short*)(ws + 3 * GATE_N * KDIM * 2);     // 1008*256 bf16
  float* bsum = (float*)(ws + 3 * GATE_N * KDIM * 2 + FC_N * KDIM * 2);     // 3*768 f32
  float* bfcp = (float*)((char*)bsum + 3 * GATE_N * 4);                     // 1008 f32

  int total = 3 * GATE_N * KDIM + FC_N * KDIM + 3 * GATE_N + FC_N;
  prep_pack<<<(total + 255) / 256, 256, 0, stream>>>(
      W0, W1, W2, bi0, bh0, bi1, bh1, bi2, bh2, Wfc, bfc, wpk, wfcp, bsum, bfcp);

  size_t shbytes = (size_t)ROWS * AP * 2 + (size_t)ROWS * GP * 4 +
                   (size_t)ROWS * LP * 4 + ROWS * 4;   // ~123 KB of the 320 KB WGP LDS
  lstm_gen<<<2, BLOCK, shbytes, stream>>>(x, emb, wpk, wfcp, bsum, bfcp, out);
}
